// SVLAE_Net_74328704024663
// MI455X (gfx1250) — matrix operand order repaired and checked by reading, compiled, not run on hardware
//
#include <hip/hip_runtime.h>
#include <stdint.h>

// ---------------------------------------------------------------------------
// Types for CDNA5 WMMA (wave32): v_wmma_f32_16x16x32_bf16
// ---------------------------------------------------------------------------
typedef __attribute__((ext_vector_type(16))) __bf16 v16bf;
typedef __attribute__((ext_vector_type(8)))  float  v8f;
typedef __attribute__((ext_vector_type(4)))  int    v4i;

union FragU { uint32_t u[8]; uint4 q[2]; v16bf v; };

// A-matrix (16x32 bf16) K-base for VGPR v, lane-group g (per ISA 7.12.2):
// V0: K=0,1 / 8,9 ; V1: 2,3 / 10,11 ; ... ; V4: 16,17 / 24,25 ; ...
__device__ __forceinline__ int a_kb(int v, int g) {
  return (v < 4) ? (2 * v + 8 * g) : (16 + 2 * (v - 4) + 8 * g);
}

__device__ __forceinline__ v8f wmma_bf16(v16bf a, v16bf b, v8f c) {
  return __builtin_amdgcn_wmma_f32_16x16x32_bf16(false, a, false, b, (short)0, c,
                                                 false, false);
}

// B fragment from pre-packed weights:
// layout index = ((nt*KC + kc)*32 + lane)*8 + v  (uint32 = 2 bf16, K even/odd)
__device__ __forceinline__ v16bf load_bfrag(const uint32_t* __restrict__ pack,
                                            int nt, int KC, int kc, int lane) {
  FragU f;
  const uint4* p = (const uint4*)(pack + ((size_t)(nt * KC + kc) * 32 + lane) * 8);
  f.q[0] = p[0];
  f.q[1] = p[1];
  return f.v;
}

// A fragment from an f32 LDS row (converted to bf16 in regs)
__device__ __forceinline__ v16bf afrag_f32(const float* row, int g, int kc) {
  v16bf a;
#pragma unroll
  for (int v = 0; v < 8; ++v) {
    int kb = a_kb(v, g);
    float2 f2 = *(const float2*)(row + kc * 32 + kb);
    a[2 * v]     = (__bf16)f2.x;
    a[2 * v + 1] = (__bf16)f2.y;
  }
  return a;
}

// A fragment directly from a bf16 LDS row
__device__ __forceinline__ v16bf afrag_bf16(const __bf16* row, int g, int kc) {
  v16bf a;
#pragma unroll
  for (int v = 0; v < 8; ++v) {
    int kb = a_kb(v, g);
    a[2 * v]     = row[kc * 32 + kb];
    a[2 * v + 1] = row[kc * 32 + kb + 1];
  }
  return a;
}

__device__ __forceinline__ float sigf(float x) { return 1.f / (1.f + __expf(-x)); }
// stable tanh: 1 - 2/(e^{2x}+1)
__device__ __forceinline__ float tanh_s(float x) { return 1.f - 2.f / (__expf(2.f * x) + 1.f); }
__device__ __forceinline__ float clip5(float x) { return fminf(fmaxf(x, -5.f), 5.f); }

// ---------------------------------------------------------------------------
// Async Global->LDS copy (gfx1250 GLOBAL_LOAD_ASYNC_TO_LDS_B128, ASYNCcnt).
// Builtin signature (from clang diagnostic): v4i addrspace(1)* src,
// v4i addrspace(3)* dst, imm offset, imm cpol.
// ---------------------------------------------------------------------------
#if __has_builtin(__builtin_amdgcn_global_load_async_to_lds_b128)
#define HAVE_ASYNC_LDS 1
typedef v4i __attribute__((address_space(1))) * gas_v4i_ptr;
typedef v4i __attribute__((address_space(3))) * las_v4i_ptr;
__device__ __forceinline__ void async_copy_b128(const float* gsrc, float* ldst) {
  __builtin_amdgcn_global_load_async_to_lds_b128(
      (gas_v4i_ptr)(gsrc), (las_v4i_ptr)(ldst), 0, 0);
}
#else
#define HAVE_ASYNC_LDS 0
#endif

__device__ __forceinline__ void wait_async_zero() {
#if __has_builtin(__builtin_amdgcn_s_wait_asynccnt)
  __builtin_amdgcn_s_wait_asynccnt(0);
#else
  asm volatile("s_wait_asynccnt 0" ::: "memory");
#endif
}

// ---------------------------------------------------------------------------
// Weight pre-pack: f32 [Nrows, ldw] (col offset) -> bf16 B-fragment layout,
// zero padded to NT*16 rows and KC*32 K.
// ---------------------------------------------------------------------------
__global__ __launch_bounds__(256) void prepack_kernel(
    const float* __restrict__ W, int ldw, int coloff, int Nrows, int K,
    uint32_t* __restrict__ out, int NT, int KC) {
  int idx = blockIdx.x * 256 + threadIdx.x;
  int total = NT * KC * 256;
  if (idx >= total) return;
  int v = idx & 7;
  int lane = (idx >> 3) & 31;
  int fc = idx >> 8;
  int kc = fc % KC;
  int nt = fc / KC;
  int n = nt * 16 + (lane & 15);
  int g = lane >> 4;
  int k = kc * 32 + a_kb(v, g);  // B 32x16: same K order as A (V0: K=0,1 / 8,9 ; ... ; V4: 16,17 / 24,25 ; ...)
  float f0 = (n < Nrows && k < K) ? W[(size_t)n * ldw + coloff + k] : 0.f;
  float f1 = (n < Nrows && k + 1 < K) ? W[(size_t)n * ldw + coloff + k + 1] : 0.f;
  union { __bf16 h[2]; uint32_t u; } pk;
  pk.h[0] = (__bf16)f0;
  pk.h[1] = (__bf16)f1;
  out[idx] = pk.u;
}

// ---------------------------------------------------------------------------
// Big GEMM: Y[M, ldy(=N)] (+)= X[M(.rev), ldx] @ Wpack^T + bias
// Block = 256 thr (8 waves), tile 64(M) x 128(N); grid = (M/64, N/128)
// X tile staged into LDS with async Global->LDS B128 copies on the aligned
// fast path (tracked by ASYNCcnt), scalar guarded path only for the K tail.
// ---------------------------------------------------------------------------
__global__ __launch_bounds__(256) void gemm_xw_kernel(
    const float* __restrict__ X, int ldx, const uint32_t* __restrict__ Bpack,
    const float* __restrict__ bias, float* __restrict__ Y, int ldy,
    int Ktot, int KC, int rev, int Tdim, int Bdim, int accmode) {
  __shared__ float Xs[64][36];  // 144B rows: 16B-aligned for async B128 stores
  const int tid = threadIdx.x;
  const int w = tid >> 5, lane = tid & 31;
  const int m = lane & 15, g2 = lane >> 4;
  const int m0 = blockIdx.x * 64;
  const int mt = w & 3;
  const int ntl0 = (w >> 2) * 4;

  v8f acc[4];
  const v8f Z = {0.f, 0.f, 0.f, 0.f, 0.f, 0.f, 0.f, 0.f};
#pragma unroll
  for (int i = 0; i < 4; ++i) acc[i] = Z;

  // loader mapping: each thread loads 8 consecutive floats of one row
  const int lrow = tid & 63, cq = tid >> 6;
  int mrow = m0 + lrow;
  if (rev) {
    int t = mrow / Bdim, b = mrow - t * Bdim;
    mrow = (Tdim - 1 - t) * Bdim + b;
  }
  const float* xrow = X + (size_t)mrow * ldx;

  for (int kc = 0; kc < KC; ++kc) {
    int c0 = kc * 32 + cq * 8;
    if (c0 + 8 <= Ktot) {  // uniform per wave (cq is wave-uniform)
#if HAVE_ASYNC_LDS
      async_copy_b128(xrow + c0, &Xs[lrow][cq * 8]);
      async_copy_b128(xrow + c0 + 4, &Xs[lrow][cq * 8 + 4]);
      wait_async_zero();
#else
      float4 f0 = *(const float4*)(xrow + c0);
      float4 f1 = *(const float4*)(xrow + c0 + 4);
      Xs[lrow][cq * 8 + 0] = f0.x; Xs[lrow][cq * 8 + 1] = f0.y;
      Xs[lrow][cq * 8 + 2] = f0.z; Xs[lrow][cq * 8 + 3] = f0.w;
      Xs[lrow][cq * 8 + 4] = f1.x; Xs[lrow][cq * 8 + 5] = f1.y;
      Xs[lrow][cq * 8 + 6] = f1.z; Xs[lrow][cq * 8 + 7] = f1.w;
#endif
    } else {
#pragma unroll
      for (int i = 0; i < 8; ++i)
        Xs[lrow][cq * 8 + i] = (c0 + i < Ktot) ? xrow[c0 + i] : 0.f;
    }
    __syncthreads();
    v16bf a = afrag_f32(&Xs[mt * 16 + m][0], g2, 0);
#pragma unroll
    for (int i = 0; i < 4; ++i) {
      int ntg = blockIdx.y * 8 + ntl0 + i;
      acc[i] = wmma_bf16(a, load_bfrag(Bpack, ntg, KC, kc, lane), acc[i]);
    }
    __syncthreads();
  }

#pragma unroll
  for (int i = 0; i < 4; ++i) {
    int ncol = blockIdx.y * 128 + (ntl0 + i) * 16 + m;
    float bsv = bias ? bias[ncol] : 0.f;
#pragma unroll
    for (int r = 0; r < 8; ++r) {
      int row = m0 + mt * 16 + 8 * g2 + r;
      size_t off = (size_t)row * ldy + ncol;
      float v = acc[i][r] + bsv;
      if (accmode) v += Y[off];
      Y[off] = v;
    }
  }
}

// ---------------------------------------------------------------------------
// Persistent bidirectional GRU encoder scan. grid.x = 2 (dir), block = 256.
// Dynamic LDS: h double buffer 2 * 128 * 132 f32 = 135168 bytes.
// Writes clipped g [T,B,256] (bwd direction time-reversed + col offset 128).
// ---------------------------------------------------------------------------
__global__ __launch_bounds__(256) void enc_scan_kernel(
    const float* __restrict__ xp_f, const float* __restrict__ xp_b,
    const uint32_t* __restrict__ packf, const uint32_t* __restrict__ packb,
    const float* __restrict__ bhhf, const float* __restrict__ bhhb,
    const float* __restrict__ enc_init, float* __restrict__ g_out) {
  constexpr int T = 500, B = 128, E = 128, H3 = 384, HS = 132;
  extern __shared__ float hbuf[];  // 2 * B * HS
  const int dir = blockIdx.x;
  const float* xp = dir ? xp_b : xp_f;
  const uint32_t* pack = dir ? packb : packf;
  const float* bhh = dir ? bhhb : bhhf;
  const int tid = threadIdx.x, w = tid >> 5, lane = tid & 31;
  const int m = lane & 15, g2 = lane >> 4;
  const v8f Z = {0.f, 0.f, 0.f, 0.f, 0.f, 0.f, 0.f, 0.f};

  for (int i = tid; i < B * E; i += 256)
    hbuf[(i / E) * HS + (i % E)] = enc_init[dir * E + (i % E)];
  __syncthreads();

  int cur = 0;
#pragma unroll 1
  for (int t = 0; t < T; ++t) {
    float* hc = hbuf + cur * B * HS;
    float* hn = hbuf + (cur ^ 1) * B * HS;
    const float* xpt = xp + (size_t)t * B * H3;
    const int tout = dir ? (T - 1 - t) : t;

    // wave w owns M-tile mt = w: hoist the 4 A fragments of h
    v16bf afr[4];
#pragma unroll
    for (int kc = 0; kc < 4; ++kc) afr[kc] = afrag_f32(hc + (w * 16 + m) * HS, g2, kc);

#pragma unroll 1
    for (int jt = 0; jt < 8; ++jt) {
      v8f ar = Z, az = Z, an = Z;
#pragma unroll
      for (int kc = 0; kc < 4; ++kc) {
        ar = wmma_bf16(afr[kc], load_bfrag(pack, jt, 4, kc, lane), ar);
        az = wmma_bf16(afr[kc], load_bfrag(pack, jt + 8, 4, kc, lane), az);
        an = wmma_bf16(afr[kc], load_bfrag(pack, jt + 16, 4, kc, lane), an);
      }
      int j = jt * 16 + m;
      float brv = bhh[j], bzv = bhh[E + j], bnv = bhh[2 * E + j];
#pragma unroll
      for (int r = 0; r < 8; ++r) {
        int b = w * 16 + 8 * g2 + r;
        const float* xr = xpt + (size_t)b * H3;
        float rr = sigf(xr[j] + ar[r] + brv);
        float zz = sigf(xr[E + j] + az[r] + bzv);
        float nh = an[r] + bnv;
        float nn = tanh_s(xr[2 * E + j] + rr * nh);
        float hold = hc[b * HS + j];
        float hv = (1.f - zz) * nn + zz * hold;
        hn[b * HS + j] = hv;
        g_out[((size_t)tout * B + b) * (2 * E) + dir * E + j] = clip5(hv);
      }
    }
    __syncthreads();
    cur ^= 1;
  }
}

// ---------------------------------------------------------------------------
// Persistent controller scan + latent sample + spike generator + AR1 calcium.
// grid = 1 block, 256 threads. Dynamic LDS:
//   h double buffer 2*128*132 f32 | genin 128*100 bf16 | cal 128*300 f32
// ---------------------------------------------------------------------------
__global__ __launch_bounds__(256) void ctrl_scan_kernel(
    const float* __restrict__ cp, const uint32_t* __restrict__ whh,
    const float* __restrict__ bhh, const float* __restrict__ ctrl_init,
    const uint32_t* __restrict__ wmu, const float* __restrict__ bmu,
    const uint32_t* __restrict__ wlv, const float* __restrict__ blv,
    const float* __restrict__ eps, const float* __restrict__ factors,
    const uint32_t* __restrict__ wspk, const float* __restrict__ bspk,
    const float* __restrict__ gainp, const float* __restrict__ biaspp,
    const float* __restrict__ logtaup, float* __restrict__ out) {
  constexpr int T = 500, B = 128, C = 128, H3 = 384, U = 64, F = 4, N = 300;
  constexpr int HS = 132, GS = 100;
  extern __shared__ char smem[];
  float* hbuf = (float*)smem;                                    // 2*B*HS f32
  __bf16* genin = (__bf16*)(smem + 2 * B * HS * 4);              // B*GS bf16
  float* cal = (float*)(smem + 2 * B * HS * 4 + B * GS * 2);     // B*N f32

  const int tid = threadIdx.x, w = tid >> 5, lane = tid & 31;
  const int m = lane & 15, g2 = lane >> 4;
  const v8f Z = {0.f, 0.f, 0.f, 0.f, 0.f, 0.f, 0.f, 0.f};

  const float gain = gainp[0], biasp = biaspp[0];
  const float decay = 1.f - __expf(-logtaup[0]);

  for (int i = tid; i < B * C; i += 256) hbuf[(i / C) * HS + (i % C)] = ctrl_init[i % C];
  for (int i = tid; i < B * N; i += 256) cal[i] = 0.f;
  for (int i = tid; i < B * GS; i += 256) genin[i] = (__bf16)0.f;
  __syncthreads();

  int cur = 0;
#pragma unroll 1
  for (int t = 0; t < T; ++t) {
    float* hc = hbuf + cur * B * HS;
    float* hn = hbuf + (cur ^ 1) * B * HS;
    const float* cpt = cp + (size_t)t * B * H3;

    // ---- phase 1: GRU with clip ----
    v16bf afr[4];
#pragma unroll
    for (int kc = 0; kc < 4; ++kc) afr[kc] = afrag_f32(hc + (w * 16 + m) * HS, g2, kc);
#pragma unroll 1
    for (int jt = 0; jt < 8; ++jt) {
      v8f ar = Z, az = Z, an = Z;
#pragma unroll
      for (int kc = 0; kc < 4; ++kc) {
        ar = wmma_bf16(afr[kc], load_bfrag(whh, jt, 4, kc, lane), ar);
        az = wmma_bf16(afr[kc], load_bfrag(whh, jt + 8, 4, kc, lane), az);
        an = wmma_bf16(afr[kc], load_bfrag(whh, jt + 16, 4, kc, lane), an);
      }
      int j = jt * 16 + m;
      float brv = bhh[j], bzv = bhh[C + j], bnv = bhh[2 * C + j];
#pragma unroll
      for (int r = 0; r < 8; ++r) {
        int b = w * 16 + 8 * g2 + r;
        const float* xr = cpt + (size_t)b * H3;
        float rr = sigf(xr[j] + ar[r] + brv);
        float zz = sigf(xr[C + j] + az[r] + bzv);
        float nn = tanh_s(xr[2 * C + j] + rr * (an[r] + bnv));
        float hold = hc[b * HS + j];
        hn[b * HS + j] = clip5((1.f - zz) * nn + zz * hold);
      }
    }
    __syncthreads();

    // ---- phase 2: mu/lv -> u -> genin (bf16), plus factors ----
    v16bf afr2[4];
#pragma unroll
    for (int kc = 0; kc < 4; ++kc) afr2[kc] = afrag_f32(hn + (w * 16 + m) * HS, g2, kc);
#pragma unroll 1
    for (int c = 0; c < 4; ++c) {
      v8f am = Z, al = Z;
#pragma unroll
      for (int kc = 0; kc < 4; ++kc) {
        am = wmma_bf16(afr2[kc], load_bfrag(wmu, c, 4, kc, lane), am);
        al = wmma_bf16(afr2[kc], load_bfrag(wlv, c, 4, kc, lane), al);
      }
      int uidx = c * 16 + m;
      float bmv = bmu[uidx], blvv = blv[uidx];
#pragma unroll
      for (int r = 0; r < 8; ++r) {
        int b = w * 16 + 8 * g2 + r;
        float ev = eps[((size_t)t * B + b) * U + uidx];
        float uu = (am[r] + bmv) + __expf(0.5f * (al[r] + blvv)) * ev;
        genin[b * GS + uidx] = (__bf16)uu;
      }
    }
    for (int i = tid; i < B * F; i += 256) {
      int b = i >> 2, q = i & 3;
      genin[b * GS + U + q] = (__bf16)factors[((size_t)t * B + b) * F + q];
    }
    __syncthreads();

    // ---- phase 3: spike GEMM + AR1 calcium ----
#pragma unroll 1
    for (int i = 0; i < 19; ++i) {
      int flat = i * 8 + w;
      int mt = flat / 19, nt = flat % 19;
      v8f acc = Z;
#pragma unroll
      for (int kc = 0; kc < 3; ++kc)
        acc = wmma_bf16(afrag_bf16(genin + (mt * 16 + m) * GS, g2, kc),
                        load_bfrag(wspk, nt, 3, kc, lane), acc);
      int n = nt * 16 + m;
      if (n < N) {
        float bs = bspk[n];
#pragma unroll
        for (int r = 0; r < 8; ++r) {
          int b = mt * 16 + 8 * g2 + r;
          float s = fmaxf(__expf(acc[r] + bs) - 1.f, 0.f);
          float cv = cal[b * N + n] * decay + gain * s + biasp;
          cal[b * N + n] = cv;
          out[((size_t)t * B + b) * N + n] = cv;
        }
      }
    }
    __syncthreads();
    cur ^= 1;
  }
}

// ---------------------------------------------------------------------------
// Host launcher
// ---------------------------------------------------------------------------
extern "C" void kernel_launch(void* const* d_in, const int* in_sizes, int n_in,
                              void* d_out, int out_size, void* d_ws, size_t ws_size,
                              hipStream_t stream) {
  (void)in_sizes; (void)n_in; (void)out_size; (void)ws_size;
  constexpr int T = 500, Bb = 128, N = 300, E = 128, C = 128, U = 64, F = 4;
  constexpr int H3 = 384;

  const float* x        = (const float*)d_in[0];
  const float* factors  = (const float*)d_in[1];
  const float* eps      = (const float*)d_in[2];
  const float* Wih_f    = (const float*)d_in[3];
  const float* Whh_f    = (const float*)d_in[4];
  const float* bih_f    = (const float*)d_in[5];
  const float* bhh_f    = (const float*)d_in[6];
  const float* Wih_b    = (const float*)d_in[7];
  const float* Whh_b    = (const float*)d_in[8];
  const float* bih_b    = (const float*)d_in[9];
  const float* bhh_b    = (const float*)d_in[10];
  const float* enc_init = (const float*)d_in[11];
  const float* cWih     = (const float*)d_in[12];
  const float* cWhh     = (const float*)d_in[13];
  const float* cbih     = (const float*)d_in[14];
  const float* cbhh     = (const float*)d_in[15];
  const float* c_init   = (const float*)d_in[16];
  const float* W_mu     = (const float*)d_in[17];
  const float* b_mu     = (const float*)d_in[18];
  const float* W_lv     = (const float*)d_in[19];
  const float* b_lv     = (const float*)d_in[20];
  const float* W_spk    = (const float*)d_in[21];
  const float* b_spk    = (const float*)d_in[22];
  const float* gainp    = (const float*)d_in[23];
  const float* biaspp   = (const float*)d_in[24];
  const float* logtaup  = (const float*)d_in[25];

  char* ws = (char*)d_ws;
  size_t off = 0;
  auto alloc = [&](size_t bytes) { char* p = ws + off; off += (bytes + 255) & ~(size_t)255; return p; };

  float* xp_f = (float*)alloc((size_t)T * Bb * H3 * 4);
  float* xp_b = (float*)alloc((size_t)T * Bb * H3 * 4);
  float* cpb  = (float*)alloc((size_t)T * Bb * H3 * 4);
  float* gbuf = (float*)alloc((size_t)T * Bb * 2 * E * 4);

  uint32_t* pk_wihf = (uint32_t*)alloc(24 * 10 * 256 * 4);
  uint32_t* pk_wihb = (uint32_t*)alloc(24 * 10 * 256 * 4);
  uint32_t* pk_wx   = (uint32_t*)alloc(24 * 10 * 256 * 4);
  uint32_t* pk_wg   = (uint32_t*)alloc(24 * 8 * 256 * 4);
  uint32_t* pk_whhf = (uint32_t*)alloc(24 * 4 * 256 * 4);
  uint32_t* pk_whhb = (uint32_t*)alloc(24 * 4 * 256 * 4);
  uint32_t* pk_whhc = (uint32_t*)alloc(24 * 4 * 256 * 4);
  uint32_t* pk_wmu  = (uint32_t*)alloc(4 * 4 * 256 * 4);
  uint32_t* pk_wlv  = (uint32_t*)alloc(4 * 4 * 256 * 4);
  uint32_t* pk_wspk = (uint32_t*)alloc(19 * 3 * 256 * 4);

  auto pp = [&](const float* W, int ldw, int co, int Nr, int K, uint32_t* o, int NT, int KC) {
    int total = NT * KC * 256;
    prepack_kernel<<<(total + 255) / 256, 256, 0, stream>>>(W, ldw, co, Nr, K, o, NT, KC);
  };
  pp(Wih_f, N, 0, H3, N, pk_wihf, 24, 10);
  pp(Wih_b, N, 0, H3, N, pk_wihb, 24, 10);
  pp(cWih, 2 * E + N, 2 * E, H3, N, pk_wx, 24, 10);    // x part of ctrl_Wih
  pp(cWih, 2 * E + N, 0, H3, 2 * E, pk_wg, 24, 8);     // g part of ctrl_Wih
  pp(Whh_f, E, 0, H3, E, pk_whhf, 24, 4);
  pp(Whh_b, E, 0, H3, E, pk_whhb, 24, 4);
  pp(cWhh, C, 0, H3, C, pk_whhc, 24, 4);
  pp(W_mu, C, 0, U, C, pk_wmu, 4, 4);
  pp(W_lv, C, 0, U, C, pk_wlv, 4, 4);
  pp(W_spk, U + F, 0, N, U + F, pk_wspk, 19, 3);

  // Phase 1: big input-projection GEMMs (M = 64000)
  dim3 gg(T * Bb / 64, H3 / 128);
  gemm_xw_kernel<<<gg, 256, 0, stream>>>(x, N, pk_wihf, bih_f, xp_f, H3, N, 10, 0, T, Bb, 0);
  gemm_xw_kernel<<<gg, 256, 0, stream>>>(x, N, pk_wihb, bih_b, xp_b, H3, N, 10, 1, T, Bb, 0);
  gemm_xw_kernel<<<gg, 256, 0, stream>>>(x, N, pk_wx, cbih, cpb, H3, N, 10, 0, T, Bb, 0);

  // Phase 2: persistent encoder scans (fwd + bwd in parallel blocks)
  enc_scan_kernel<<<2, 256, 2 * 128 * 132 * 4, stream>>>(
      xp_f, xp_b, pk_whhf, pk_whhb, bhh_f, bhh_b, enc_init, gbuf);

  // Phase 3: cp += g @ Wg^T
  gemm_xw_kernel<<<gg, 256, 0, stream>>>(gbuf, 2 * E, pk_wg, nullptr, cpb, H3, 2 * E, 8, 0, T, Bb, 1);

  // Phase 4: persistent controller scan + generator
  size_t smem = (size_t)2 * 128 * 132 * 4 + 128 * 100 * 2 + 128 * 300 * 4;  // 314368 B
  ctrl_scan_kernel<<<1, 256, smem, stream>>>(
      cpb, pk_whhc, cbhh, c_init, pk_wmu, b_mu, pk_wlv, b_lv, eps, factors,
      pk_wspk, b_spk, gainp, biaspp, logtaup, (float*)d_out);
}
